// MultiHeadAttention_66297115181497
// MI455X (gfx1250) — matrix operand order repaired and checked by reading, not compiled, not run on hardware
//
#include <hip/hip_runtime.h>
#include <hip/hip_bf16.h>

#define DEV __device__ __forceinline__

typedef __attribute__((ext_vector_type(16))) __bf16 v16bf;
typedef __attribute__((ext_vector_type(8)))  __bf16 v8bf;
typedef __attribute__((ext_vector_type(8)))  float  v8f;
typedef __attribute__((ext_vector_type(4)))  unsigned int u32x4;
typedef __attribute__((ext_vector_type(4)))  int i32x4;
typedef __attribute__((ext_vector_type(8)))  int i32x8;

static constexpr int B_  = 4;
static constexpr int S_  = 2048;
static constexpr int D_  = 1024;
static constexpr int H_  = 16;
static constexpr int DK_ = 64;
static constexpr int M_  = B_ * S_;   // 8192 total rows

// -------- workspace layout (bytes), all offsets 16B aligned --------
static constexpr size_t XB_OFF = 0;                                  // x bf16   [8192,1024]
static constexpr size_t WQ_OFF = XB_OFF + (size_t)M_ * D_ * 2;       // wq bf16  [1024,1024]
static constexpr size_t WK_OFF = WQ_OFF + (size_t)D_ * D_ * 2;
static constexpr size_t WV_OFF = WK_OFF + (size_t)D_ * D_ * 2;
static constexpr size_t WO_OFF = WV_OFF + (size_t)D_ * D_ * 2;
static constexpr size_t Q_OFF  = WO_OFF + (size_t)D_ * D_ * 2;       // Q bf16 [B,H,S,64]
static constexpr size_t K_OFF  = Q_OFF  + (size_t)M_ * D_ * 2;       // K bf16 [B,H,S,64]
static constexpr size_t V_OFF  = K_OFF  + (size_t)M_ * D_ * 2;       // Vt bf16 [B,H,64,S]
static constexpr size_t O_OFF  = V_OFF  + (size_t)M_ * D_ * 2;       // attn out bf16 [B,S,D]
// total = 92,274,688 bytes

#if __has_builtin(__builtin_amdgcn_tensor_load_to_lds)
#define HAVE_TDM 1
#else
#define HAVE_TDM 0
#endif

// float -> bf16 bits, round-to-nearest-even
DEV unsigned short f2bf(float f) {
  unsigned int u = __builtin_bit_cast(unsigned int, f);
  unsigned int rnd = 0x7fffu + ((u >> 16) & 1u);
  return (unsigned short)((u + rnd) >> 16);
}

// assemble a 16-element bf16 fragment from two aligned 16B chunks
DEV v16bf load_frag(const unsigned short* p0, const unsigned short* p1) {
  v16bf r;
  *reinterpret_cast<v8bf*>(&r)       = *reinterpret_cast<const v8bf*>(p0);
  *(reinterpret_cast<v8bf*>(&r) + 1) = *reinterpret_cast<const v8bf*>(p1);
  return r;
}

DEV v8f wmma_bf16(v16bf a, v16bf b, v8f c) {
  return __builtin_amdgcn_wmma_f32_16x16x32_bf16(
      /*neg_a=*/false, a, /*neg_b=*/false, b,
      /*c_mod=*/(short)0, c, /*reuse_a=*/false, /*reuse_b=*/false);
}

#if HAVE_TDM
// TDM: DMA a 2D bf16 tile (tile_d0 contiguous elems per row, tile_d1 rows,
// rows separated by row_stride elems) from global memory into LDS.
// D# bitfields per CDNA5 ISA.
DEV void tdm_load_2d(unsigned lds_byte_addr, const void* gptr,
                     unsigned tile_d0, unsigned tile_d1,
                     unsigned long long row_stride_elems) {
  unsigned long long ga = (unsigned long long)(uintptr_t)gptr;
  u32x4 g0;
  g0.x = 1u;                                           // count=1, user descriptor
  g0.y = lds_byte_addr;                                // lds_addr (bytes)
  g0.z = (unsigned)ga;                                 // global_addr[31:0]
  g0.w = (unsigned)((ga >> 32) & 0x01ffffffu) | (2u << 30);  // addr[56:32] | type=2
  i32x8 g1;
  g1[0] = (int)(1u << 16);                             // wg_mask=0, data_size=1 (2B)
  g1[1] = (int)((tile_d0 & 0xffffu) << 16);            // tensor_dim0[15:0]
  g1[2] = (int)(((tile_d0 >> 16) & 0xffffu) |          // tensor_dim0[31:16]
                ((tile_d1 & 0xffffu) << 16));          // tensor_dim1[15:0]
  g1[3] = (int)(((tile_d1 >> 16) & 0xffffu) |          // tensor_dim1[31:16]
                ((tile_d0 & 0xffffu) << 16));          // tile_dim0
  g1[4] = (int)(tile_d1 & 0xffffu);                    // tile_dim1 (tile_dim2=0)
  g1[5] = (int)(row_stride_elems & 0xffffffffu);       // tensor_dim0_stride[31:0]
  g1[6] = (int)((row_stride_elems >> 32) & 0xffffu);   // stride[47:32] (dim1_stride=0)
  g1[7] = 0;
  i32x4 gz = {0, 0, 0, 0};
#if __clang_major__ >= 23
  i32x8 gz8 = {0, 0, 0, 0, 0, 0, 0, 0};
  __builtin_amdgcn_tensor_load_to_lds(g0, g1, gz, gz, gz8, 0);
#else
  __builtin_amdgcn_tensor_load_to_lds(g0, g1, gz, gz, 0);
#endif
}
#endif  // HAVE_TDM

// ---------------------------------------------------------------------------
// Kernel 0: f32 -> bf16 conversion, 4 elements / thread
// ---------------------------------------------------------------------------
__global__ void __launch_bounds__(256)
cvt_f32_to_bf16_kernel(const float* __restrict__ in,
                       unsigned short* __restrict__ out, int n) {
  int i = (blockIdx.x * blockDim.x + threadIdx.x) * 4;
  if (i + 3 < n) {
    const float4 v = *reinterpret_cast<const float4*>(in + i);
    out[i + 0] = f2bf(v.x);
    out[i + 1] = f2bf(v.y);
    out[i + 2] = f2bf(v.z);
    out[i + 3] = f2bf(v.w);
  }
}

// ---------------------------------------------------------------------------
// Kernel 1: fused QKV projection  y = x @ W^T
// grid (M/32, D/64, 3), block 32 (one wave). Wave computes 32x64 f32 tile.
// z=0 -> Q [B,H,S,64]; z=1 -> K [B,H,S,64]; z=2 -> V transposed [B,H,64,S]
// ---------------------------------------------------------------------------
__global__ void __launch_bounds__(32)
qkv_gemm_kernel(const unsigned short* __restrict__ Xb,
                const unsigned short* __restrict__ Wq,
                const unsigned short* __restrict__ Wk,
                const unsigned short* __restrict__ Wv,
                unsigned short* __restrict__ Qo,
                unsigned short* __restrict__ Ko,
                unsigned short* __restrict__ Vo) {
  const int m0 = blockIdx.x * 32;
  const int n0 = blockIdx.y * 64;
  const int z  = blockIdx.z;
  const unsigned short* W = (z == 0) ? Wq : (z == 1) ? Wk : Wv;

  const int lane = threadIdx.x;
  const int ln   = lane & 15;
  const int hi   = lane >> 4;

  v8f acc[2][4];
#pragma unroll
  for (int i = 0; i < 2; ++i)
#pragma unroll
    for (int j = 0; j < 4; ++j)
#pragma unroll
      for (int r = 0; r < 8; ++r) acc[i][j][r] = 0.0f;

  for (int k0 = 0; k0 < D_; k0 += 32) {
    v16bf a[2], b[4];
#pragma unroll
    for (int i = 0; i < 2; ++i) {
      const unsigned short* p = Xb + (size_t)(m0 + 16 * i + ln) * D_ + k0 + 8 * hi;
      a[i] = load_frag(p, p + 16);
    }
#pragma unroll
    for (int j = 0; j < 4; ++j) {
      const unsigned short* p = W + (size_t)(n0 + 16 * j + ln) * D_ + k0 + 8 * hi;
      b[j] = load_frag(p, p + 16);
    }
#pragma unroll
    for (int i = 0; i < 2; ++i)
#pragma unroll
      for (int j = 0; j < 4; ++j)
        acc[i][j] = wmma_bf16(a[i], b[j], acc[i][j]);
  }

#pragma unroll
  for (int i = 0; i < 2; ++i) {
#pragma unroll
    for (int j = 0; j < 4; ++j) {
      const int n = n0 + 16 * j + ln;
      const int h = n >> 6, d = n & 63;
#pragma unroll
      for (int r = 0; r < 8; ++r) {
        const int M  = m0 + 16 * i + r + 8 * hi;
        const int bb = M >> 11, s = M & 2047;
        const unsigned short v = f2bf(acc[i][j][r]);
        if (z == 0)
          Qo[(((size_t)bb * H_ + h) * S_ + s) * DK_ + d] = v;
        else if (z == 1)
          Ko[(((size_t)bb * H_ + h) * S_ + s) * DK_ + d] = v;
        else
          Vo[(((size_t)bb * H_ + h) * DK_ + d) * S_ + s] = v;
      }
    }
  }
}

// ---------------------------------------------------------------------------
// Kernel 2: causal flash attention.
// grid (S/64, B*H), block 128 (4 waves). Each wave owns a 16-row q-tile.
// K/V tiles for each 32-key step are staged in LDS by the Tensor Data Mover
// (double buffered, TENSORcnt-synchronized), shared by all 4 waves.
// ---------------------------------------------------------------------------
__global__ void __launch_bounds__(128)
flash_attn_kernel(const unsigned short* __restrict__ Q,
                  const unsigned short* __restrict__ K,
                  const unsigned short* __restrict__ Vt,
                  unsigned short* __restrict__ O) {
  const int q0b = blockIdx.x * 64;
  const int bh  = blockIdx.y;
  const int tid  = threadIdx.x;
  const int w    = tid >> 5;        // wave in block
  const int lane = tid & 31;
  const int ln   = lane & 15;
  const int hi   = lane >> 4;
  const int q0   = q0b + w * 16;    // this wave's q-tile base

  const unsigned short* Qh = Q  + (size_t)bh * S_ * DK_;
  const unsigned short* Kh = K  + (size_t)bh * S_ * DK_;
  const unsigned short* Vh = Vt + (size_t)bh * DK_ * S_;

  __shared__ alignas(16) unsigned short Ktile[2][32 * 64];  // [key][d]
  __shared__ alignas(16) unsigned short Vtile[2][64 * 32];  // [d][kv]
  __shared__ alignas(16) unsigned short Plds[4][16 * 32];   // per-wave P tile

  const float NEG_INF = -__builtin_inff();
  const float scale   = 0.125f;     // 1/sqrt(64)

  // Q A-frags (16 x 64 over two 16x32 fragments), resident for whole kernel
  v16bf qa[2];
#pragma unroll
  for (int t = 0; t < 2; ++t) {
    const unsigned short* p = Qh + (size_t)(q0 + ln) * DK_ + 32 * t + 8 * hi;
    qa[t] = load_frag(p, p + 16);
  }

  v8f o[4];
  float mrow[8], lrow[8];
#pragma unroll
  for (int r = 0; r < 8; ++r) { mrow[r] = NEG_INF; lrow[r] = 0.0f; }
#pragma unroll
  for (int c = 0; c < 4; ++c)
#pragma unroll
    for (int r = 0; r < 8; ++r) o[c][r] = 0.0f;

  const int ntiles = (q0b >> 5) + 2;   // covers keys 0 .. q0b+63

#if HAVE_TDM
  if (w == 0) {   // prologue: stage tile 0 into buffer 0
    tdm_load_2d((unsigned)(uintptr_t)&Ktile[0][0], Kh, 64, 32, 64);
    tdm_load_2d((unsigned)(uintptr_t)&Vtile[0][0], Vh, 32, 64, 2048);
  }
#endif

  for (int it = 0; it < ntiles; ++it) {
    const int kt  = it * 32;
    const int cur = it & 1;

#if HAVE_TDM
    if (w == 0) __builtin_amdgcn_s_wait_tensorcnt(0);  // tile `it` landed in LDS
    __syncthreads();                                   // publish; also WAR fence
    if (w == 0 && it + 1 < ntiles) {                   // prefetch next tile via TDM
      const int nk = kt + 32;
      tdm_load_2d((unsigned)(uintptr_t)&Ktile[cur ^ 1][0], Kh + (size_t)nk * DK_, 64, 32, 64);
      tdm_load_2d((unsigned)(uintptr_t)&Vtile[cur ^ 1][0], Vh + nk, 32, 64, 2048);
    }
#else
    __syncthreads();                                   // WAR fence
    // cooperative synchronous staging fallback
    for (int off = tid * 8; off < 32 * 64; off += 128 * 8)
      *reinterpret_cast<v8bf*>(&Ktile[cur][off]) =
          *reinterpret_cast<const v8bf*>(Kh + (size_t)kt * DK_ + off);
    for (int off = tid * 8; off < 64 * 32; off += 128 * 8)
      *reinterpret_cast<v8bf*>(&Vtile[cur][off]) =
          *reinterpret_cast<const v8bf*>(Vh + (size_t)(off >> 5) * S_ + kt + (off & 31));
    __syncthreads();
#endif

    if (kt <= q0 + 15) {   // wave-uniform causal bound for this q-tile
      const unsigned short* Kt  = &Ktile[cur][0];
      const unsigned short* Vtl = &Vtile[cur][0];

      // ---- S = Q @ K^T for 32 keys (two 16x16 C tiles) ----
      v8f s0, s1;
#pragma unroll
      for (int r = 0; r < 8; ++r) { s0[r] = 0.0f; s1[r] = 0.0f; }
#pragma unroll
      for (int t = 0; t < 2; ++t) {
        const unsigned short* p0 = Kt + (size_t)ln * 64        + 32 * t + 8 * hi;
        const unsigned short* p1 = Kt + (size_t)(16 + ln) * 64 + 32 * t + 8 * hi;
        v16bf kb0 = load_frag(p0, p0 + 16);
        v16bf kb1 = load_frag(p1, p1 + 16);
        s0 = wmma_bf16(qa[t], kb0, s0);
        s1 = wmma_bf16(qa[t], kb1, s1);
      }

      // ---- scale + causal mask + online softmax ----
      const bool needMask = (kt + 31 > q0);
#pragma unroll
      for (int r = 0; r < 8; ++r) {
        float v0 = s0[r] * scale;
        float v1 = s1[r] * scale;
        if (needMask) {
          const int qrow = q0 + r + 8 * hi;
          v0 = (kt + ln      > qrow) ? NEG_INF : v0;
          v1 = (kt + 16 + ln > qrow) ? NEG_INF : v1;
        }
        float rm = fmaxf(v0, v1);
#pragma unroll
        for (int off = 1; off < 16; off <<= 1)
          rm = fmaxf(rm, __shfl_xor(rm, off, 32));
        const float mnew  = fmaxf(mrow[r], rm);
        const float alpha = __expf(mrow[r] - mnew);
        mrow[r] = mnew;
        const float p0e = __expf(v0 - mnew);
        const float p1e = __expf(v1 - mnew);
        float rs = p0e + p1e;
#pragma unroll
        for (int off = 1; off < 16; off <<= 1)
          rs += __shfl_xor(rs, off, 32);
        lrow[r] = lrow[r] * alpha + rs;
#pragma unroll
        for (int c = 0; c < 4; ++c) o[c][r] *= alpha;
        s0[r] = p0e;
        s1[r] = p1e;
      }

      // ---- P: C-layout -> A-frag layout via per-wave LDS round trip ----
      unsigned short* pw = &Plds[w][0];
#pragma unroll
      for (int r = 0; r < 8; ++r) {
        pw[(r + 8 * hi) * 32 + ln]      = f2bf(s0[r]);
        pw[(r + 8 * hi) * 32 + 16 + ln] = f2bf(s1[r]);
      }
      // same-wave LDS ops are in-order; fence HW counter + compiler ordering
      asm volatile("s_wait_dscnt 0x0" ::: "memory");
      const unsigned short* pp = pw + ln * 32 + 8 * hi;
      v16bf pa = load_frag(pp, pp + 16);

      // ---- O += P @ V ----
#pragma unroll
      for (int c = 0; c < 4; ++c) {
        const unsigned short* vp = Vtl + (size_t)(16 * c + ln) * 32 + 8 * hi;
        v16bf vb = load_frag(vp, vp + 16);
        o[c] = wmma_bf16(pa, vb, o[c]);
      }
    }
  }

  // ---- epilogue: normalize and store attn output bf16 [B,S,D] ----
  const int bb = bh >> 4, h = bh & 15;
#pragma unroll
  for (int r = 0; r < 8; ++r) {
    const float inv = 1.0f / lrow[r];
    const int s = q0 + r + 8 * hi;
#pragma unroll
    for (int c = 0; c < 4; ++c) {
      O[((size_t)bb * S_ + s) * D_ + h * DK_ + 16 * c + ln] = f2bf(o[c][r] * inv);
    }
  }
}

// ---------------------------------------------------------------------------
// Kernel 3: output projection  out = O @ w_o^T + b_o  (f32 out)
// grid (M/32, D/64), block 32.
// ---------------------------------------------------------------------------
__global__ void __launch_bounds__(32)
oproj_gemm_kernel(const unsigned short* __restrict__ Ob,
                  const unsigned short* __restrict__ Wo,
                  const float* __restrict__ bias,
                  float* __restrict__ out) {
  const int m0 = blockIdx.x * 32;
  const int n0 = blockIdx.y * 64;
  const int lane = threadIdx.x;
  const int ln   = lane & 15;
  const int hi   = lane >> 4;

  v8f acc[2][4];
#pragma unroll
  for (int i = 0; i < 2; ++i)
#pragma unroll
    for (int j = 0; j < 4; ++j)
#pragma unroll
      for (int r = 0; r < 8; ++r) acc[i][j][r] = 0.0f;

  for (int k0 = 0; k0 < D_; k0 += 32) {
    v16bf a[2], b[4];
#pragma unroll
    for (int i = 0; i < 2; ++i) {
      const unsigned short* p = Ob + (size_t)(m0 + 16 * i + ln) * D_ + k0 + 8 * hi;
      a[i] = load_frag(p, p + 16);
    }
#pragma unroll
    for (int j = 0; j < 4; ++j) {
      const unsigned short* p = Wo + (size_t)(n0 + 16 * j + ln) * D_ + k0 + 8 * hi;
      b[j] = load_frag(p, p + 16);
    }
#pragma unroll
    for (int i = 0; i < 2; ++i)
#pragma unroll
      for (int j = 0; j < 4; ++j)
        acc[i][j] = wmma_bf16(a[i], b[j], acc[i][j]);
  }

#pragma unroll
  for (int i = 0; i < 2; ++i) {
#pragma unroll
    for (int j = 0; j < 4; ++j) {
      const int n  = n0 + 16 * j + ln;
      const float bv = bias[n];
#pragma unroll
      for (int r = 0; r < 8; ++r) {
        const int M = m0 + 16 * i + r + 8 * hi;
        out[(size_t)M * D_ + n] = acc[i][j][r] + bv;
      }
    }
  }
}

// ---------------------------------------------------------------------------
extern "C" void kernel_launch(void* const* d_in, const int* in_sizes, int n_in,
                              void* d_out, int out_size, void* d_ws, size_t ws_size,
                              hipStream_t stream) {
  (void)in_sizes; (void)n_in; (void)out_size; (void)ws_size;

  const float* x  = (const float*)d_in[0];
  const float* wq = (const float*)d_in[1];
  const float* wk = (const float*)d_in[2];
  const float* wv = (const float*)d_in[3];
  const float* wo = (const float*)d_in[4];
  const float* bo = (const float*)d_in[5];
  float* out = (float*)d_out;

  char* ws = (char*)d_ws;
  unsigned short* xb  = (unsigned short*)(ws + XB_OFF);
  unsigned short* wqb = (unsigned short*)(ws + WQ_OFF);
  unsigned short* wkb = (unsigned short*)(ws + WK_OFF);
  unsigned short* wvb = (unsigned short*)(ws + WV_OFF);
  unsigned short* wob = (unsigned short*)(ws + WO_OFF);
  unsigned short* Qb  = (unsigned short*)(ws + Q_OFF);
  unsigned short* Kb  = (unsigned short*)(ws + K_OFF);
  unsigned short* Vb  = (unsigned short*)(ws + V_OFF);
  unsigned short* Ob  = (unsigned short*)(ws + O_OFF);

  // f32 -> bf16 conversions
  cvt_f32_to_bf16_kernel<<<dim3(M_ * D_ / 1024), 256, 0, stream>>>(x,  xb,  M_ * D_);
  cvt_f32_to_bf16_kernel<<<dim3(D_ * D_ / 1024), 256, 0, stream>>>(wq, wqb, D_ * D_);
  cvt_f32_to_bf16_kernel<<<dim3(D_ * D_ / 1024), 256, 0, stream>>>(wk, wkb, D_ * D_);
  cvt_f32_to_bf16_kernel<<<dim3(D_ * D_ / 1024), 256, 0, stream>>>(wv, wvb, D_ * D_);
  cvt_f32_to_bf16_kernel<<<dim3(D_ * D_ / 1024), 256, 0, stream>>>(wo, wob, D_ * D_);

  // fused QKV projection
  qkv_gemm_kernel<<<dim3(M_ / 32, D_ / 64, 3), 32, 0, stream>>>(
      xb, wqb, wkb, wvb, Qb, Kb, Vb);

  // causal flash attention (TDM-staged K/V tiles, 4 waves / block)
  flash_attn_kernel<<<dim3(S_ / 64, B_ * H_), 128, 0, stream>>>(Qb, Kb, Vb, Ob);

  // output projection + bias
  oproj_gemm_kernel<<<dim3(M_ / 32, D_ / 64), 32, 0, stream>>>(Ob, wob, bo, out);
}
